// Decoder_24438363914871
// MI455X (gfx1250) — matrix operand order repaired and checked by reading, not compiled, not run on hardware
//
#include <hip/hip_runtime.h>
#include <hip/hip_bf16.h>

// Problem dims (fixed by the reference)
#define T_STEPS 2048
#define BATCH   256
#define IDIM    64
#define HDIM    256
#define BW      16   // batch rows per workgroup (== WMMA M)
#define NWG     (BATCH / BW)

typedef __attribute__((ext_vector_type(16))) __bf16 v16bf;
typedef __attribute__((ext_vector_type(8)))  __bf16 bf16x8;
typedef __attribute__((ext_vector_type(8)))  float  v8f;
typedef __attribute__((ext_vector_type(4)))  float  f32x4;
typedef __attribute__((ext_vector_type(4)))  int    i32x4;

#if __has_builtin(__builtin_amdgcn_global_load_async_to_lds_b128) && \
    __has_builtin(__builtin_amdgcn_s_wait_asynccnt)
#define XSTAGE_ASYNC 1
typedef __attribute__((address_space(1))) i32x4 gi32x4_t;
typedef __attribute__((address_space(3))) i32x4 li32x4_t;
#else
#define XSTAGE_ASYNC 0
#endif

__device__ __forceinline__ v8f wmma_bf16(v16bf a, v16bf b, v8f c) {
    // 8 args: (neg_a, A, neg_b, B, c_mod, C, reuse_a, reuse_b)
    return __builtin_amdgcn_wmma_f32_16x16x32_bf16(false, a, false, b, (short)0, c, false, false);
}

__device__ __forceinline__ v16bf cat8(bf16x8 a, bf16x8 b) {
    return __builtin_shufflevector(a, b, 0,1,2,3,4,5,6,7,8,9,10,11,12,13,14,15);
}

__device__ __forceinline__ v8f splat8(float x) {
    v8f v;
    #pragma unroll
    for (int i = 0; i < 8; ++i) v[i] = x;
    return v;
}

// B fragment (32x16): B[k][n] = W[n0+lo][kb*32 + k']; per-lane K = hi*8 + {0..7} and 16 + hi*8 + {0..7}
__device__ __forceinline__ v16bf load_bfrag(const __bf16* __restrict__ W, int ldk,
                                            int n0, int kb, int lo, int hi) {
    const __bf16* p = W + (size_t)(n0 + lo) * ldk + kb * 32 + hi * 8;
    bf16x8 a = *(const bf16x8*)(p);
    bf16x8 b = *(const bf16x8*)(p + 16);
    return cat8(a, b);
}

// A fragment (16x32) from row-major bf16 LDS S[16][ldk]:
// per-lane row M = lo; K = kb*32 + hi*8 + {0..7} and kb*32 + 16 + hi*8 + {0..7}
__device__ __forceinline__ v16bf load_afrag_lds(const __bf16* S, int ldk,
                                                int kb, int lo, int hi) {
    const __bf16* p = S + lo * ldk + kb * 32 + hi * 8;
    bf16x8 a = *(const bf16x8*)(p);
    bf16x8 b = *(const bf16x8*)(p + 16);
    return cat8(a, b);
}

// A fragment (16x32) from row-major fp32 LDS, converting to bf16 in regs
__device__ __forceinline__ v16bf load_afrag_lds_f32(const float* S, int ldk,
                                                    int kb, int lo, int hi) {
    const float* p = S + lo * ldk + kb * 32 + hi * 8;
    f32x4 a0 = *(const f32x4*)(p);
    f32x4 a1 = *(const f32x4*)(p + 4);
    f32x4 b0 = *(const f32x4*)(p + 16);
    f32x4 b1 = *(const f32x4*)(p + 20);
    v16bf r;
    #pragma unroll
    for (int i = 0; i < 4; ++i) {
        r[i]      = (__bf16)a0[i];
        r[i + 4]  = (__bf16)a1[i];
        r[i + 8]  = (__bf16)b0[i];
        r[i + 12] = (__bf16)b1[i];
    }
    return r;
}

__device__ __forceinline__ float sigf(float x) { return 1.0f / (1.0f + __expf(-x)); }

// ---------------- prep: fp32 weights -> bf16 in workspace ----------------
__global__ void cvt_weights(const float* __restrict__ Wih, const float* __restrict__ Whh,
                            const float* __restrict__ Wout,
                            __bf16* __restrict__ oih, __bf16* __restrict__ ohh,
                            __bf16* __restrict__ oout) {
    int i = blockIdx.x * blockDim.x + threadIdx.x;
    if (i < 768 * 64)  oih[i]  = (__bf16)Wih[i];
    if (i < 768 * 256) ohh[i]  = (__bf16)Whh[i];
    if (i < 64 * 256)  oout[i] = (__bf16)Wout[i];
}

// ---------------- fused persistent GRU decoder ----------------
__global__ __launch_bounds__(256)
void gru_fused(const float* __restrict__ input, const float* __restrict__ hidden,
               const float* __restrict__ W_dec, const float* __restrict__ b_dec,
               const float* __restrict__ b_ih, const float* __restrict__ b_hh,
               const float* __restrict__ b_out,
               const __bf16* __restrict__ Wih, const __bf16* __restrict__ Whh,
               const __bf16* __restrict__ Wout,
               float* __restrict__ out) {
    __shared__ __attribute__((aligned(16))) float  sh [BW][HDIM];  // fp32 h state
    __shared__ __attribute__((aligned(16))) __bf16 shb[BW][HDIM];  // bf16 mirror (A operand)
    __shared__ __attribute__((aligned(16))) float  sx [BW][IDIM];  // fp32 x_t tile

    const int tid  = threadIdx.x;
    const int w    = tid >> 5;          // wave id (wave32)
    const int lane = tid & 31;
    const int lo   = lane & 15;
    const int hi   = lane >> 4;
    const int b0   = blockIdx.x * BW;   // batch base for this WG
    const int jb   = w * 32;            // this wave's hidden-unit slice [jb, jb+32)

    // ---- h0 = hidden @ W_dec^T + b_dec ----
    for (int e = tid; e < BW * HDIM; e += 256) {
        int m = e >> 8, j = e & 255;
        const float* hb = hidden + (size_t)(b0 + m) * 3;
        float v = b_dec[j] + hb[0] * W_dec[j * 3 + 0]
                           + hb[1] * W_dec[j * 3 + 1]
                           + hb[2] * W_dec[j * 3 + 2];
        sh[m][j]  = v;
        shb[m][j] = (__bf16)v;
    }

    // ---- per-lane gate biases (gate cols c0 = jb+lo, c1 = jb+16+lo) ----
    const int c0 = jb + lo, c1 = jb + 16 + lo;
    const float br0 = b_ih[c0] + b_hh[c0];
    const float br1 = b_ih[c1] + b_hh[c1];
    const float bz0 = b_ih[256 + c0] + b_hh[256 + c0];
    const float bz1 = b_ih[256 + c1] + b_hh[256 + c1];
    const float bi0 = b_ih[512 + c0], bi1 = b_ih[512 + c1];
    const float bh0 = b_hh[512 + c0], bh1 = b_hh[512 + c1];
    const float bo  = (w < 4) ? b_out[w * 16 + lo] : 0.0f;

    __syncthreads();

    v16bf hf[8];

    for (int t = 0; t <= T_STEPS; ++t) {
        // ---- stage x_t early so the DMA overlaps the output GEMM ----
        // x_t = (t==0 ? 0 : input[t-1]); wave w stages rows 2w, 2w+1
        if (t < T_STEPS) {
            const int row = (w << 1) | hi;
            const int cc  = lo * 4;
            if (t == 0) {
                f32x4 z4 = {0.f, 0.f, 0.f, 0.f};
                *(f32x4*)&sx[row][cc] = z4;
            } else {
                const float* gp = input + ((size_t)(t - 1) * BATCH + (b0 + row)) * IDIM + cc;
#if XSTAGE_ASYNC
                __builtin_amdgcn_global_load_async_to_lds_b128(
                    (gi32x4_t*)gp, (li32x4_t*)&sx[row][cc], 0, 0);
#else
                f32x4 v4 = *(const f32x4*)gp;
                *(f32x4*)&sx[row][cc] = v4;
#endif
            }
        }

        // h A-fragments = state after step t-1 (also feeds output GEMM of t-1)
        #pragma unroll
        for (int kb = 0; kb < 8; ++kb)
            hf[kb] = load_afrag_lds(&shb[0][0], HDIM, kb, lo, hi);

        // ---- output GEMM for step t-1: out[b, t-1, :] = hs[t-1] @ Wout^T + b_out ----
        if (w < 4 && t > 0) {
            v8f ao = splat8(bo);   // bias pre-seeded in accumulator
            #pragma unroll
            for (int kb = 0; kb < 8; ++kb)
                ao = wmma_bf16(hf[kb], load_bfrag(Wout, HDIM, w * 16, kb, lo, hi), ao);
            #pragma unroll
            for (int v = 0; v < 8; ++v) {
                int b = b0 + v + 8 * hi;
                out[((size_t)b * T_STEPS + (t - 1)) * IDIM + w * 16 + lo] = ao[v];
            }
        }
        if (t == T_STEPS) break;

#if XSTAGE_ASYNC
        if (t > 0) __builtin_amdgcn_s_wait_asynccnt(0);
#endif
        __syncthreads();

        v16bf ax0 = load_afrag_lds_f32(&sx[0][0], IDIM, 0, lo, hi);
        v16bf ax1 = load_afrag_lds_f32(&sx[0][0], IDIM, 1, lo, hi);

        // accumulators seeded with the gate biases (bias[col] is v-invariant)
        v8f ar0 = splat8(br0), ar1 = splat8(br1);
        v8f az0 = splat8(bz0), az1 = splat8(bz1);
        v8f ai0 = splat8(bi0), ai1 = splat8(bi1);
        v8f ah0 = splat8(bh0), ah1 = splat8(bh1);

        // ---- gi = x_t @ Wih^T  (K = 64) ----
        #pragma unroll
        for (int kb = 0; kb < 2; ++kb) {
            v16bf a = kb ? ax1 : ax0;
            ar0 = wmma_bf16(a, load_bfrag(Wih, IDIM, jb,            kb, lo, hi), ar0);
            ar1 = wmma_bf16(a, load_bfrag(Wih, IDIM, jb + 16,       kb, lo, hi), ar1);
            az0 = wmma_bf16(a, load_bfrag(Wih, IDIM, 256 + jb,      kb, lo, hi), az0);
            az1 = wmma_bf16(a, load_bfrag(Wih, IDIM, 256 + jb + 16, kb, lo, hi), az1);
            ai0 = wmma_bf16(a, load_bfrag(Wih, IDIM, 512 + jb,      kb, lo, hi), ai0);
            ai1 = wmma_bf16(a, load_bfrag(Wih, IDIM, 512 + jb + 16, kb, lo, hi), ai1);
        }
        // ---- gh = h @ Whh^T  (K = 256) ----
        #pragma unroll
        for (int kb = 0; kb < 8; ++kb) {
            ar0 = wmma_bf16(hf[kb], load_bfrag(Whh, HDIM, jb,            kb, lo, hi), ar0);
            ar1 = wmma_bf16(hf[kb], load_bfrag(Whh, HDIM, jb + 16,       kb, lo, hi), ar1);
            az0 = wmma_bf16(hf[kb], load_bfrag(Whh, HDIM, 256 + jb,      kb, lo, hi), az0);
            az1 = wmma_bf16(hf[kb], load_bfrag(Whh, HDIM, 256 + jb + 16, kb, lo, hi), az1);
            ah0 = wmma_bf16(hf[kb], load_bfrag(Whh, HDIM, 512 + jb,      kb, lo, hi), ah0);
            ah1 = wmma_bf16(hf[kb], load_bfrag(Whh, HDIM, 512 + jb + 16, kb, lo, hi), ah1);
        }

        // ---- gates + state update (lane-local; C/D elem (v): row M = v+8*hi, col = lo) ----
        #pragma unroll
        for (int v = 0; v < 8; ++v) {
            int m = v + 8 * hi;
            {
                float r  = sigf(ar0[v]);
                float z  = sigf(az0[v]);
                float n  = tanhf(ai0[v] + r * ah0[v]);
                float ho = sh[m][c0];
                float hn = n + z * (ho - n);
                sh[m][c0]  = hn;
                shb[m][c0] = (__bf16)hn;
            }
            {
                float r  = sigf(ar1[v]);
                float z  = sigf(az1[v]);
                float n  = tanhf(ai1[v] + r * ah1[v]);
                float ho = sh[m][c1];
                float hn = n + z * (ho - n);
                sh[m][c1]  = hn;
                shb[m][c1] = (__bf16)hn;
            }
        }
        __syncthreads();
    }
}

extern "C" void kernel_launch(void* const* d_in, const int* in_sizes, int n_in,
                              void* d_out, int out_size, void* d_ws, size_t ws_size,
                              hipStream_t stream) {
    const float* input  = (const float*)d_in[0];
    const float* hidden = (const float*)d_in[1];
    const float* W_dec  = (const float*)d_in[2];
    const float* b_dec  = (const float*)d_in[3];
    const float* W_ih   = (const float*)d_in[4];
    const float* W_hh   = (const float*)d_in[5];
    const float* b_ih   = (const float*)d_in[6];
    const float* b_hh   = (const float*)d_in[7];
    const float* W_out  = (const float*)d_in[8];
    const float* b_out  = (const float*)d_in[9];
    float* out = (float*)d_out;

    // bf16 weight copies in workspace: 768*64 + 768*256 + 64*256 bf16 = 512 KB
    __bf16* wih_b  = (__bf16*)d_ws;
    __bf16* whh_b  = wih_b + 768 * 64;
    __bf16* wout_b = whh_b + 768 * 256;

    cvt_weights<<<768, 256, 0, stream>>>(W_ih, W_hh, W_out, wih_b, whh_b, wout_b);
    gru_fused<<<NWG, 256, 0, stream>>>(input, hidden, W_dec, b_dec, b_ih, b_hh, b_out,
                                       wih_b, whh_b, wout_b, out);
}
